// UNONdPoints_80255758893086
// MI455X (gfx1250) — matrix operand order repaired and checked by reading, not compiled, not run on hardware
//
#include <hip/hip_runtime.h>
#include <math.h>

#define BB    2
#define NPTS  4096
#define WID   128
#define HID   128
#define KNN   16
#define KPAD  160   // 131 padded to 5 slabs of 32
#define NB    12

// padded LDS row strides (f16 units): odd multiples of 8 -> bank-conflict-free
// fragment loads (lane bank offset becomes 4*l mod 64, a full permutation).
#define LDA   168   // edge A tile (KPAD=160 cols + pad)
#define LDW   136   // 128-wide weight tiles
#define LDH   136   // 128-wide hidden tiles
#define FLDA  136   // ff input tile
#define FLDW1 264   // ff W1 rows are 256 wide
#define FLDH  264   // ff hidden (256 wide)
#define FLDW2 136
#define FLDC  132   // f32 staging stride (floats)

typedef _Float16 f16;
typedef __attribute__((ext_vector_type(16))) _Float16 v16h;
typedef __attribute__((ext_vector_type(8)))  float    v8f;
typedef int vi4 __attribute__((__vector_size__(4 * sizeof(int))));

#if defined(__has_builtin)
#if __has_builtin(__builtin_amdgcn_global_load_async_to_lds_b128) && \
    __has_builtin(__builtin_amdgcn_s_wait_asynccnt)
#define USE_ASYNC_LDS 1
#endif
#endif

#if defined(USE_ASYNC_LDS)
typedef __attribute__((address_space(1))) vi4 gvi4;
typedef __attribute__((address_space(3))) vi4 lvi4;
#endif

__device__ __forceinline__ float gelu_f(float x) {
  return 0.5f * x * (1.0f + erff(x * 0.70710678118654752f));
}

// ---- WMMA fragment loads from LDS ------------------------------------------
// A (16x32 f16, MxK): lane l: row = l&15, half=l>>4.
// VGPR0..3 = K[8h .. 8h+7], VGPR4..7 = K[16+8h .. 16+8h+7]  (ISA 7.12.2)
__device__ __forceinline__ v16h load_a_frag(const f16* p0, int lda) {
  const int l = threadIdx.x & 31;
  const f16* p = p0 + (l & 15) * lda + (l >> 4) * 8;
  v16h r;
  ((float4*)&r)[0] = *(const float4*)(p);
  ((float4*)&r)[1] = *(const float4*)(p + 16);
  return r;
}
// B (32x16 f16, KxN): lane l: col n = l&15, half h=l>>4;
// elems 0..7 = K[8h .. 8h+7], elems 8..15 = K[16+8h .. 16+8h+7] (matches A).
__device__ __forceinline__ v16h load_b_frag(const f16* p0, int ldb) {
  const int l = threadIdx.x & 31;
  const f16* p = p0 + (l & 15) + (l >> 4) * 8 * ldb;
  v16h r;
#pragma unroll
  for (int i = 0; i < 8; ++i) {
    r[i] = p[i * ldb];
    r[8 + i] = p[(16 + i) * ldb];
  }
  return r;
}
#define WMMA_F16(A, Bf, C) \
  __builtin_amdgcn_wmma_f32_16x16x32_f16(false, (A), false, (Bf), (short)0, (C), false, false)

// Copy a rows x cols (f16, cols % 8 == 0) weight matrix from global into LDS
// with padded row stride ldst. Uses the CDNA5 async global->LDS path when the
// toolchain declares the builtin; otherwise a plain uint4 copy.
__device__ __forceinline__ void copy_w(const f16* __restrict__ g, f16* s,
                                       int rows, int cols, int ldst, int tid) {
  const int cpr = cols >> 3;          // 16B chunks per row
  const int total = rows * cpr;
#if defined(USE_ASYNC_LDS)
  for (int i = tid; i < total; i += 256) {
    const int r = i / cpr, c = i - r * cpr;
    __builtin_amdgcn_global_load_async_to_lds_b128(
        (gvi4*)(g + (size_t)r * cols) + c,
        (lvi4*)(s + r * ldst) + c, 0, 0);
  }
  __builtin_amdgcn_s_wait_asynccnt(0);
#else
  for (int i = tid; i < total; i += 256) {
    const int r = i / cpr, c = i - r * cpr;
    ((uint4*)(s + r * ldst))[c] = ((const uint4*)(g + (size_t)r * cols))[c];
  }
#endif
}

// ---- weight conversion ------------------------------------------------------
__global__ void cvt_f16_kernel(const float* __restrict__ in, f16* __restrict__ out, int n) {
  int i = blockIdx.x * 256 + threadIdx.x;
  if (i < n) out[i] = (f16)in[i];
}
// kW1: (NB,131,128) -> (NB,160,128) zero-padded rows
__global__ void cvt_w1_kernel(const float* __restrict__ in, f16* __restrict__ out) {
  int i = blockIdx.x * 256 + threadIdx.x;
  if (i >= NB * KPAD * HID) return;
  int blk = i / (KPAD * HID);
  int rem = i - blk * (KPAD * HID);
  int r = rem / HID, c = rem - r * HID;
  out[i] = (r < 131) ? (f16)in[(blk * 131 + r) * HID + c] : (f16)0.f;
}

// ---- brute force kNN (per level coords, K=16 incl self) --------------------
__global__ void knn_kernel(const float* __restrict__ coords, int* __restrict__ out, int Nl) {
  __shared__ float sc[256 * 3];
  const int b = blockIdx.y;
  const int q = blockIdx.x * 256 + threadIdx.x;
  const float* cb = coords + (size_t)b * Nl * 3;
  const float qx = cb[q*3+0], qy = cb[q*3+1], qz = cb[q*3+2];
  float dk[KNN]; int ik[KNN];
#pragma unroll
  for (int i = 0; i < KNN; ++i) { dk[i] = 3.4e38f; ik[i] = 0; }
  for (int t = 0; t < Nl; t += 256) {
    const int j = t + threadIdx.x;
    sc[threadIdx.x*3+0] = cb[j*3+0];
    sc[threadIdx.x*3+1] = cb[j*3+1];
    sc[threadIdx.x*3+2] = cb[j*3+2];
    __syncthreads();
    for (int jj = 0; jj < 256; ++jj) {
      const float dx = qx - sc[jj*3+0], dy = qy - sc[jj*3+1], dz = qz - sc[jj*3+2];
      const float d = dx*dx + dy*dy + dz*dz;
      if (d < dk[KNN-1]) {
        dk[KNN-1] = d; ik[KNN-1] = t + jj;
#pragma unroll
        for (int s = KNN-1; s > 0; --s) {
          if (dk[s] < dk[s-1]) {
            float td = dk[s]; dk[s] = dk[s-1]; dk[s-1] = td;
            int ti = ik[s]; ik[s] = ik[s-1]; ik[s-1] = ti;
          }
        }
      }
    }
    __syncthreads();
  }
  int* o = out + ((size_t)b * Nl + q) * KNN;
#pragma unroll
  for (int i = 0; i < KNN; ++i) o[i] = ik[i];
}

// ---- nearest coarse point for unpool ---------------------------------------
__global__ void nn_up_kernel(const float* __restrict__ cf, const float* __restrict__ cc,
                             int* __restrict__ out, int Nf, int Nc) {
  __shared__ float sc[256 * 3];
  const int b = blockIdx.y;
  const int q = blockIdx.x * 256 + threadIdx.x;
  const float* f = cf + (size_t)b * Nf * 3;
  const float* c = cc + (size_t)b * Nc * 3;
  const float qx = f[q*3+0], qy = f[q*3+1], qz = f[q*3+2];
  float best = 3.4e38f; int bi = 0;
  for (int t = 0; t < Nc; t += 256) {
    const int j = t + threadIdx.x;
    sc[threadIdx.x*3+0] = c[j*3+0];
    sc[threadIdx.x*3+1] = c[j*3+1];
    sc[threadIdx.x*3+2] = c[j*3+2];
    __syncthreads();
    for (int jj = 0; jj < 256; ++jj) {
      const float dx = qx - sc[jj*3+0], dy = qy - sc[jj*3+1], dz = qz - sc[jj*3+2];
      const float d = dx*dx + dy*dy + dz*dz;
      if (d < best) { best = d; bi = t + jj; }
    }
    __syncthreads();
  }
  out[(size_t)b * Nf + q] = bi;
}

// ---- gathers / unpool -------------------------------------------------------
__global__ void gather_feat_kernel(const float* __restrict__ in, const int* __restrict__ idx,
                                   float* __restrict__ out, int Nin, int Nout) {
  const int t = blockIdx.x * 256 + threadIdx.x;    // (b*Nout + i)*32 + c4
  const int c4 = t & 31;
  const int row = t >> 5;
  const int b = row / Nout;
  const int src = idx[row];
  ((float4*)out)[(size_t)row * 32 + c4] =
      ((const float4*)in)[((size_t)b * Nin + src) * 32 + c4];
}
__global__ void gather_coords_kernel(const float* __restrict__ in, const int* __restrict__ idx,
                                     float* __restrict__ out, int Nin, int Nout) {
  const int t = blockIdx.x * 256 + threadIdx.x;
  if (t >= BB * Nout * 3) return;
  const int d = t % 3;
  const int row = t / 3;
  const int b = row / Nout;
  const int src = idx[row];
  out[(size_t)row * 3 + d] = in[((size_t)b * Nin + src) * 3 + d];
}
__global__ void unpool_add_kernel(const float* __restrict__ coarse, const int* __restrict__ nn,
                                  float* __restrict__ skip_io, int Nf, int Nc) {
  const int t = blockIdx.x * 256 + threadIdx.x;
  const int c4 = t & 31;
  const int row = t >> 5;
  const int b = row / Nf;
  const int src = nn[row];
  float4 a = ((const float4*)coarse)[((size_t)b * Nc + src) * 32 + c4];
  float4 s = ((float4*)skip_io)[(size_t)row * 32 + c4];
  s.x += a.x; s.y += a.y; s.z += a.z; s.w += a.w;
  ((float4*)skip_io)[(size_t)row * 32 + c4] = s;
}

// ---- lift: [x,coords](6) -> 128 -> 128 -------------------------------------
__global__ void lift_kernel(const float* __restrict__ x, const float* __restrict__ coords,
                            const float* __restrict__ lW1, const float* __restrict__ lb1,
                            const float* __restrict__ lW2, const float* __restrict__ lb2,
                            float* __restrict__ h) {
  const int p = blockIdx.x, c = threadIdx.x;
  __shared__ float t[WID];
  __shared__ float in6[6];
  if (c < 3) { in6[c] = x[(size_t)p*3 + c]; in6[3+c] = coords[(size_t)p*3 + c]; }
  __syncthreads();
  float a = lb1[c];
#pragma unroll
  for (int k = 0; k < 6; ++k) a += in6[k] * lW1[k*WID + c];
  t[c] = gelu_f(a);
  __syncthreads();
  float o = lb2[c];
  for (int k = 0; k < WID; ++k) o += t[k] * lW2[k*WID + c];
  h[(size_t)p*WID + c] = o;
}

// ---- projection: 128 -> 128 -> 1 -------------------------------------------
__global__ void proj_kernel(const float* __restrict__ h,
                            const float* __restrict__ pW1, const float* __restrict__ pb1,
                            const float* __restrict__ pW2, const float* __restrict__ pb2,
                            float* __restrict__ out) {
  const int p = blockIdx.x, c = threadIdx.x;
  __shared__ float hr[WID];
  __shared__ float ps[4];
  hr[c] = h[(size_t)p*WID + c];
  __syncthreads();
  float a = pb1[c];
  for (int k = 0; k < WID; ++k) a += hr[k] * pW1[k*WID + c];
  float v = gelu_f(a) * pW2[c];
#pragma unroll
  for (int m = 16; m >= 1; m >>= 1) v += __shfl_xor(v, m, 32);
  if ((c & 31) == 0) ps[c >> 5] = v;
  __syncthreads();
  if (c == 0) out[p] = ps[0] + ps[1] + ps[2] + ps[3] + pb2[0];
}

// ---- WMMA edge-MLP block: gather + 3-layer MLP + mean + LN -----------------
// One WG = 8 query points = 128 edge rows; 8 waves, wave w owns the 16-row
// strip holding all 16 neighbors of point w. 256 threads.
__global__ __launch_bounds__(256, 1)
void edge_block_kernel(const float* __restrict__ h, const float* __restrict__ coords,
                       const int* __restrict__ nnidx,
                       const f16* __restrict__ gw1, const f16* __restrict__ gw2,
                       const f16* __restrict__ gw3,
                       const float* __restrict__ b1, const float* __restrict__ b2,
                       const float* __restrict__ b3, const float* __restrict__ opb,
                       const float* __restrict__ lnw, const float* __restrict__ lnb,
                       float* __restrict__ y, int Nl) {
  __shared__ f16 As[128 * LDA];    // 42.0KB edge rows (rel|feat|pad)
  __shared__ f16 Ws[KPAD * LDW];   // 42.5KB current weight matrix
  __shared__ f16 H1[128 * LDH];    // 34.0KB
  __shared__ f16 H2[128 * LDH];    // 34.0KB
  const int tid = threadIdx.x;
  const int lane = tid & 31;
  const int wv = tid >> 5;
  const int base = blockIdx.x * 8;
  const int boff = (base / Nl) * Nl;

  { // gather A tile
    const int r = tid >> 1, sub = tid & 1;
    const int prow = base + (r >> 4);
    const int nrow = boff + nnidx[(size_t)prow * KNN + (r & 15)];
    const float4* src = (const float4*)(h + (size_t)nrow * WID) + sub * 16;
    f16* dst = As + r * LDA + 3 + sub * 64;
#pragma unroll
    for (int i = 0; i < 16; ++i) {
      float4 v = src[i];
      dst[i*4+0] = (f16)v.x; dst[i*4+1] = (f16)v.y;
      dst[i*4+2] = (f16)v.z; dst[i*4+3] = (f16)v.w;
    }
    if (sub == 0) {
#pragma unroll
      for (int d = 0; d < 3; ++d)
        As[r*LDA + d] = (f16)(coords[(size_t)nrow*3 + d] - coords[(size_t)prow*3 + d]);
    } else {
      for (int c = 3 + WID; c < KPAD; ++c) As[r*LDA + c] = (f16)0.f;
    }
  }
  copy_w(gw1, Ws, KPAD, HID, LDW, tid);
  __syncthreads();

  { // layer 1: [128xKPAD]x[KPADx128] -> gelu -> H1 (two acc chains)
    v16h a[5];
#pragma unroll
    for (int s = 0; s < 5; ++s) a[s] = load_a_frag(As + (wv*16)*LDA + s*32, LDA);
    for (int nt = 0; nt < 8; ++nt) {
      v16h bf[5];
#pragma unroll
      for (int s = 0; s < 5; ++s) bf[s] = load_b_frag(Ws + (s*32)*LDW + nt*16, LDW);
      v8f acc0 = {}, acc1 = {};
      acc0 = WMMA_F16(a[0], bf[0], acc0);
      acc1 = WMMA_F16(a[1], bf[1], acc1);
      acc0 = WMMA_F16(a[2], bf[2], acc0);
      acc1 = WMMA_F16(a[3], bf[3], acc1);
      acc0 = WMMA_F16(a[4], bf[4], acc0);
      v8f acc = acc0 + acc1;
      const int n = nt*16 + (lane & 15);
      const float bias = b1[n];
      const int mb = wv*16 + 8*(lane >> 4);
#pragma unroll
      for (int v = 0; v < 8; ++v) H1[(mb+v)*LDH + n] = (f16)gelu_f(acc[v] + bias);
    }
  }
  __syncthreads();
  copy_w(gw2, Ws, HID, HID, LDW, tid);
  __syncthreads();

  { // layer 2 -> gelu -> H2
    v16h a[4];
#pragma unroll
    for (int s = 0; s < 4; ++s) a[s] = load_a_frag(H1 + (wv*16)*LDH + s*32, LDH);
    for (int nt = 0; nt < 8; ++nt) {
      v16h bf[4];
#pragma unroll
      for (int s = 0; s < 4; ++s) bf[s] = load_b_frag(Ws + (s*32)*LDW + nt*16, LDW);
      v8f acc0 = {}, acc1 = {};
      acc0 = WMMA_F16(a[0], bf[0], acc0);
      acc1 = WMMA_F16(a[1], bf[1], acc1);
      acc0 = WMMA_F16(a[2], bf[2], acc0);
      acc1 = WMMA_F16(a[3], bf[3], acc1);
      v8f acc = acc0 + acc1;
      const int n = nt*16 + (lane & 15);
      const float bias = b2[n];
      const int mb = wv*16 + 8*(lane >> 4);
#pragma unroll
      for (int v = 0; v < 8; ++v) H2[(mb+v)*LDH + n] = (f16)gelu_f(acc[v] + bias);
    }
  }
  __syncthreads();
  copy_w(gw3, Ws, HID, WID, LDW, tid);
  __syncthreads();

  { // layer 3 + mean over 16 neighbors + residual + LN (all within wave wv)
    v16h a[4];
#pragma unroll
    for (int s = 0; s < 4; ++s) a[s] = load_a_frag(H2 + (wv*16)*LDH + s*32, LDH);
    float msg[8];
    const int n15 = lane & 15;
    const int prow = base + wv;
#pragma unroll
    for (int nt = 0; nt < 8; ++nt) {
      v16h bf[4];
#pragma unroll
      for (int s = 0; s < 4; ++s) bf[s] = load_b_frag(Ws + (s*32)*LDW + nt*16, LDW);
      v8f acc0 = {}, acc1 = {};
      acc0 = WMMA_F16(a[0], bf[0], acc0);
      acc1 = WMMA_F16(a[1], bf[1], acc1);
      acc0 = WMMA_F16(a[2], bf[2], acc0);
      acc1 = WMMA_F16(a[3], bf[3], acc1);
      v8f acc = acc0 + acc1;
      float ssum = 0.f;
#pragma unroll
      for (int v = 0; v < 8; ++v) ssum += acc[v];      // 8 neighbor rows (this half)
      ssum += __shfl_xor(ssum, 16, 32);                // + other half -> 16 rows
      const int n = nt*16 + n15;
      msg[nt] = ssum * (1.0f/16.0f) + b3[n] + opb[n] + h[(size_t)prow*WID + n];
    }
    float s1 = 0.f, s2 = 0.f;
#pragma unroll
    for (int nt = 0; nt < 8; ++nt) { s1 += msg[nt]; s2 += msg[nt]*msg[nt]; }
#pragma unroll
    for (int m = 1; m < 16; m <<= 1) { s1 += __shfl_xor(s1, m, 32); s2 += __shfl_xor(s2, m, 32); }
    const float mean = s1 * (1.0f/128.0f);
    const float inv = rsqrtf(s2 * (1.0f/128.0f) - mean*mean + 1e-5f);
    if (lane < 16) {
#pragma unroll
      for (int nt = 0; nt < 8; ++nt) {
        const int n = nt*16 + n15;
        y[(size_t)prow*WID + n] = (msg[nt] - mean) * inv * lnw[n] + lnb[n];
      }
    }
  }
}

// ---- WMMA feed-forward block: 128->256 gelu ->128 + residual + LN ----------
// One WG = 64 rows; 8 waves = (mt 0..3) x (ns 0..1) split over output tiles.
__global__ __launch_bounds__(256, 1)
void ff_block_kernel(const float* __restrict__ y,
                     const f16* __restrict__ gw1, const f16* __restrict__ gw2,
                     const float* __restrict__ fbias1, const float* __restrict__ fbias2,
                     const float* __restrict__ lnw, const float* __restrict__ lnb,
                     float* __restrict__ hout) {
  __shared__ f16 As[64 * FLDA];     // 17.0KB
  __shared__ f16 Ws[256 * FLDW2];   // 68.0KB (also used as 128 x FLDW1)
  __shared__ f16 Hs[64 * FLDH];     // 33.0KB
  __shared__ float Cst[64 * FLDC];  // 33.0KB
  const int tid = threadIdx.x;
  const int lane = tid & 31;
  const int wv = tid >> 5;
  const int mt = wv & 3, ns = wv >> 2;
  const int rbase = blockIdx.x * 64;

  { // load y rows -> f16
    const int r = tid >> 2, sub = tid & 3;
    const float4* src = (const float4*)(y + (size_t)(rbase + r) * WID) + sub * 8;
    f16* dst = As + r * FLDA + sub * 32;
#pragma unroll
    for (int i = 0; i < 8; ++i) {
      float4 v = src[i];
      dst[i*4+0] = (f16)v.x; dst[i*4+1] = (f16)v.y;
      dst[i*4+2] = (f16)v.z; dst[i*4+3] = (f16)v.w;
    }
  }
  copy_w(gw1, Ws, WID, 256, FLDW1, tid);
  __syncthreads();

  { // layer 1: [64x128]x[128x256] -> gelu -> Hs
    v16h a[4];
#pragma unroll
    for (int s = 0; s < 4; ++s) a[s] = load_a_frag(As + (mt*16)*FLDA + s*32, FLDA);
    for (int t = 0; t < 8; ++t) {
      const int nt = ns*8 + t;
      v16h bf[4];
#pragma unroll
      for (int s = 0; s < 4; ++s) bf[s] = load_b_frag(Ws + (s*32)*FLDW1 + nt*16, FLDW1);
      v8f acc0 = {}, acc1 = {};
      acc0 = WMMA_F16(a[0], bf[0], acc0);
      acc1 = WMMA_F16(a[1], bf[1], acc1);
      acc0 = WMMA_F16(a[2], bf[2], acc0);
      acc1 = WMMA_F16(a[3], bf[3], acc1);
      v8f acc = acc0 + acc1;
      const int n = nt*16 + (lane & 15);
      const float bias = fbias1[n];
      const int mb = mt*16 + 8*(lane >> 4);
#pragma unroll
      for (int v = 0; v < 8; ++v) Hs[(mb+v)*FLDH + n] = (f16)gelu_f(acc[v] + bias);
    }
  }
  __syncthreads();
  copy_w(gw2, Ws, 256, WID, FLDW2, tid);
  __syncthreads();

  { // layer 2: [64x256]x[256x128] + bias + residual -> Cst (f32)
    v16h a[8];
#pragma unroll
    for (int s = 0; s < 8; ++s) a[s] = load_a_frag(Hs + (mt*16)*FLDH + s*32, FLDH);
    for (int t = 0; t < 4; ++t) {
      const int nt = ns*4 + t;
      v16h bf[8];
#pragma unroll
      for (int s = 0; s < 8; ++s) bf[s] = load_b_frag(Ws + (s*32)*FLDW2 + nt*16, FLDW2);
      v8f acc0 = {}, acc1 = {};
#pragma unroll
      for (int s = 0; s < 8; s += 2) {
        acc0 = WMMA_F16(a[s],   bf[s],   acc0);
        acc1 = WMMA_F16(a[s+1], bf[s+1], acc1);
      }
      v8f acc = acc0 + acc1;
      const int n = nt*16 + (lane & 15);
      const float bias = fbias2[n];
      const int mb = mt*16 + 8*(lane >> 4);
#pragma unroll
      for (int v = 0; v < 8; ++v) {
        const int row = mb + v;
        Cst[row*FLDC + n] = acc[v] + bias + y[(size_t)(rbase + row)*WID + n];
      }
    }
  }
  __syncthreads();

  { // per-row LayerNorm
    const int r = tid >> 2, sub = tid & 3;
    const float* rowp = Cst + r*FLDC + sub*32;
    float s1 = 0.f, s2 = 0.f;
#pragma unroll
    for (int c = 0; c < 32; ++c) { const float v = rowp[c]; s1 += v; s2 += v*v; }
    s1 += __shfl_xor(s1, 1, 32); s1 += __shfl_xor(s1, 2, 32);
    s2 += __shfl_xor(s2, 1, 32); s2 += __shfl_xor(s2, 2, 32);
    const float mean = s1 * (1.0f/128.0f);
    const float inv = rsqrtf(s2 * (1.0f/128.0f) - mean*mean + 1e-5f);
    float* op = hout + (size_t)(rbase + r)*WID + sub*32;
#pragma unroll
    for (int c = 0; c < 32; ++c) {
      const int n = sub*32 + c;
      op[c] = (rowp[c] - mean) * inv * lnw[n] + lnb[n];
    }
  }
}

// ---------------------------------------------------------------------------
extern "C" void kernel_launch(void* const* d_in, const int* in_sizes, int n_in,
                              void* d_out, int out_size, void* d_ws, size_t ws_size,
                              hipStream_t stream) {
  (void)in_sizes; (void)n_in; (void)out_size; (void)ws_size;
  const float* x      = (const float*)d_in[0];
  const float* coords = (const float*)d_in[1];
  const float* lW1 = (const float*)d_in[2];
  const float* lb1 = (const float*)d_in[3];
  const float* lW2 = (const float*)d_in[4];
  const float* lb2 = (const float*)d_in[5];
  const float* kW1 = (const float*)d_in[6];
  const float* kb1 = (const float*)d_in[7];
  const float* kW2 = (const float*)d_in[8];
  const float* kb2 = (const float*)d_in[9];
  const float* kW3 = (const float*)d_in[10];
  const float* kb3 = (const float*)d_in[11];
  const float* opb = (const float*)d_in[12];
  const float* fW1 = (const float*)d_in[13];
  const float* fb1 = (const float*)d_in[14];
  const float* fW2 = (const float*)d_in[15];
  const float* fb2 = (const float*)d_in[16];
  const float* lnw = (const float*)d_in[17];
  const float* lnb = (const float*)d_in[18];
  const float* pW1 = (const float*)d_in[19];
  const float* pb1 = (const float*)d_in[20];
  const float* pW2 = (const float*)d_in[21];
  const float* pb2 = (const float*)d_in[22];
  const int* pool0 = (const int*)d_in[23];
  const int* pool1 = (const int*)d_in[24];
  const int* pool2 = (const int*)d_in[25];

  const int N0 = NPTS, N1 = NPTS/2, N2 = NPTS/4, N3 = NPTS/8;

  char* wsb = (char*)d_ws;
  size_t off = 0;
  auto alloc = [&](size_t bytes) -> void* {
    void* p = wsb + off;
    off = (off + bytes + 255) & ~(size_t)255;
    return p;
  };
  float* F0 = (float*)alloc((size_t)BB*N0*WID*4);   // level-0 features / skip0
  float* Y0 = (float*)alloc((size_t)BB*N0*WID*4);
  float* F1 = (float*)alloc((size_t)BB*N1*WID*4);
  float* Y1 = (float*)alloc((size_t)BB*N1*WID*4);
  float* F2 = (float*)alloc((size_t)BB*N2*WID*4);
  float* Y2 = (float*)alloc((size_t)BB*N2*WID*4);
  float* F3 = (float*)alloc((size_t)BB*N3*WID*4);
  float* C1 = (float*)alloc((size_t)BB*N1*3*4);
  float* C2 = (float*)alloc((size_t)BB*N2*3*4);
  float* C3 = (float*)alloc((size_t)BB*N3*3*4);
  int* knn0 = (int*)alloc((size_t)BB*N0*KNN*4);
  int* knn1 = (int*)alloc((size_t)BB*N1*KNN*4);
  int* knn2 = (int*)alloc((size_t)BB*N2*KNN*4);
  int* U0 = (int*)alloc((size_t)BB*N0*4);
  int* U1 = (int*)alloc((size_t)BB*N1*4);
  int* U2 = (int*)alloc((size_t)BB*N2*4);
  f16* w1f = (f16*)alloc((size_t)NB*KPAD*HID*2);
  f16* w2f = (f16*)alloc((size_t)NB*HID*HID*2);
  f16* w3f = (f16*)alloc((size_t)NB*HID*WID*2);
  f16* f1f = (f16*)alloc((size_t)NB*WID*256*2);
  f16* f2f = (f16*)alloc((size_t)NB*256*WID*2);

  // weights -> f16 (kW1 padded to 160 rows)
  cvt_w1_kernel<<<(NB*KPAD*HID + 255)/256, 256, 0, stream>>>(kW1, w1f);
  cvt_f16_kernel<<<(NB*HID*HID + 255)/256, 256, 0, stream>>>(kW2, w2f, NB*HID*HID);
  cvt_f16_kernel<<<(NB*HID*WID + 255)/256, 256, 0, stream>>>(kW3, w3f, NB*HID*WID);
  cvt_f16_kernel<<<(NB*WID*256 + 255)/256, 256, 0, stream>>>(fW1, f1f, NB*WID*256);
  cvt_f16_kernel<<<(NB*256*WID + 255)/256, 256, 0, stream>>>(fW2, f2f, NB*256*WID);

  // lift
  lift_kernel<<<BB*N0, WID, 0, stream>>>(x, coords, lW1, lb1, lW2, lb2, F0);

  auto run_block = [&](int blk, float* F, float* Y, const float* C, const int* kn, int Nl) {
    edge_block_kernel<<<BB*Nl/8, 256, 0, stream>>>(F, C, kn,
        w1f + (size_t)blk*KPAD*HID, w2f + (size_t)blk*HID*HID, w3f + (size_t)blk*HID*WID,
        kb1 + blk*HID, kb2 + blk*HID, kb3 + blk*WID, opb + blk*WID,
        lnw + blk*WID, lnb + blk*WID, Y, Nl);
    ff_block_kernel<<<BB*Nl/64, 256, 0, stream>>>(Y,
        f1f + (size_t)blk*WID*256, f2f + (size_t)blk*256*WID,
        fb1 + blk*256, fb2 + blk*WID, lnw + blk*WID, lnb + blk*WID, F);
  };

  // ---- down path ----
  knn_kernel<<<dim3(N0/256, BB), 256, 0, stream>>>(coords, knn0, N0);
  run_block(0, F0, Y0, coords, knn0, N0);
  run_block(1, F0, Y0, coords, knn0, N0);       // F0 = skip0
  gather_feat_kernel<<<BB*N1*32/256, 256, 0, stream>>>(F0, pool0, F1, N0, N1);
  gather_coords_kernel<<<(BB*N1*3 + 255)/256, 256, 0, stream>>>(coords, pool0, C1, N0, N1);

  knn_kernel<<<dim3(N1/256, BB), 256, 0, stream>>>(C1, knn1, N1);
  run_block(2, F1, Y1, C1, knn1, N1);
  run_block(3, F1, Y1, C1, knn1, N1);           // F1 = skip1
  gather_feat_kernel<<<BB*N2*32/256, 256, 0, stream>>>(F1, pool1, F2, N1, N2);
  gather_coords_kernel<<<(BB*N2*3 + 255)/256, 256, 0, stream>>>(C1, pool1, C2, N1, N2);

  knn_kernel<<<dim3(N2/256, BB), 256, 0, stream>>>(C2, knn2, N2);
  run_block(4, F2, Y2, C2, knn2, N2);
  run_block(5, F2, Y2, C2, knn2, N2);           // F2 = skip2
  gather_feat_kernel<<<BB*N3*32/256, 256, 0, stream>>>(F2, pool2, F3, N2, N3);
  gather_coords_kernel<<<(BB*N3*3 + 255)/256, 256, 0, stream>>>(C2, pool2, C3, N2, N3);

  // ---- up path ----
  nn_up_kernel<<<dim3(N2/256, BB), 256, 0, stream>>>(C2, C3, U2, N2, N3);
  unpool_add_kernel<<<BB*N2*32/256, 256, 0, stream>>>(F3, U2, F2, N2, N3);
  run_block(10, F2, Y2, C2, knn2, N2);
  run_block(11, F2, Y2, C2, knn2, N2);

  nn_up_kernel<<<dim3(N1/256, BB), 256, 0, stream>>>(C1, C2, U1, N1, N2);
  unpool_add_kernel<<<BB*N1*32/256, 256, 0, stream>>>(F2, U1, F1, N1, N2);
  run_block(8, F1, Y1, C1, knn1, N1);
  run_block(9, F1, Y1, C1, knn1, N1);

  nn_up_kernel<<<dim3(N0/256, BB), 256, 0, stream>>>(coords, C1, U0, N0, N1);
  unpool_add_kernel<<<BB*N0*32/256, 256, 0, stream>>>(F1, U0, F0, N0, N1);
  run_block(6, F0, Y0, coords, knn0, N0);
  run_block(7, F0, Y0, coords, knn0, N0);

  // projection
  proj_kernel<<<BB*N0, WID, 0, stream>>>(F0, pW1, pb1, pW2, pb2, (float*)d_out);
}
